// sparse_MLP_51694226374949
// MI455X (gfx1250) — matrix operand order repaired and checked by reading, not compiled, not run on hardware
//
#include <hip/hip_runtime.h>

// Sizes from the reference
#define B_   8192
#define D_   256
#define H_   512
#define BT   64          // batch rows per persistent workgroup
#define NTHR 256         // 8 waves (wave32)

typedef __attribute__((ext_vector_type(16))) __bf16 v16bf;
typedef __attribute__((ext_vector_type(8)))  float  v8f;

union FragU { v16bf v; uint4 q[2]; };

// A-operand (16x32 bf16, M x K): lane l holds row M=l%16, K chunks
// [kb, kb+8) in VGPRs0-3 and [kb+16, kb+24) in VGPRs4-7 with kb=(l/16)*8.
// Row-major source => two 16B loads at +0 and +16 elements.
__device__ __forceinline__ v16bf load_a_frag(const __bf16* p) {
  FragU u;
  u.q[0] = *(const uint4*)(p);
  u.q[1] = *(const uint4*)(p + 16);
  return u.v;
}

// B-operand (32x16 bf16, K x N): lane l holds column N=l%16, K chunks
// [kb, kb+8) and [kb+16, kb+24) with kb=(l/16)*8 => two 16B loads.
__device__ __forceinline__ v16bf load_b_frag(const __bf16* p) {
  FragU u;
  u.q[0] = *(const uint4*)(p);
  u.q[1] = *(const uint4*)(p + 16);
  return u.v;
}

// ---------------- prologue: fp32 weights -> bf16 (mask folded into W1) ----
__global__ void cvt_w1_masked(const float* __restrict__ w1,
                              const int* __restrict__ mask,
                              __bf16* __restrict__ dst, int total) {
  for (int idx = blockIdx.x * blockDim.x + threadIdx.x; idx < total;
       idx += gridDim.x * blockDim.x) {
    int i = idx / (H_ * D_);
    int k = idx % D_;
    float v = w1[idx] * (float)mask[i * D_ + k];
    dst[idx] = (__bf16)v;
  }
}

__global__ void cvt_plain(const float* __restrict__ src,
                          __bf16* __restrict__ dst, int total) {
  for (int idx = blockIdx.x * blockDim.x + threadIdx.x; idx < total;
       idx += gridDim.x * blockDim.x)
    dst[idx] = (__bf16)src[idx];
}

// ---------------- persistent chained-MLP kernel ---------------------------
__global__ __launch_bounds__(NTHR) void sparse_mlp_chain(
    const float*  __restrict__ z,    // (B, D)
    const __bf16* __restrict__ w1m,  // (D, H, D) bf16, mask applied
    const __bf16* __restrict__ w2,   // (D, H, H) bf16
    const float*  __restrict__ b2,   // (D, H)
    const float*  __restrict__ w3,   // (D, H)  (W3 squeezed)
    const float*  __restrict__ b3,   // (D)
    float*        __restrict__ out)  // (B, D)
{
  __shared__ __bf16 s_inp[BT * D_];  // 32 KB: cols<i hold x, cols>=i hold z
  __shared__ __bf16 s_h1 [BT * H_];  // 64 KB: relu(fc1) activations
  __shared__ float  s_out[BT];       // fused fc3 accumulator

  const int tid  = threadIdx.x;
  const int wave = tid >> 5;
  const int lane = tid & 31;
  const int mt = wave & 3;     // M-tile (4 x 16 rows = BT)
  const int ng = wave >> 2;    // N-tile group (0/1)
  const int lh = lane >> 4;    // wave half
  const int ln = lane & 15;
  const int row0 = blockIdx.x * BT;

  for (int e = tid; e < BT * D_; e += NTHR)
    s_inp[e] = (__bf16)z[(size_t)(row0 + e / D_) * D_ + (e % D_)];
  if (tid < BT) s_out[tid] = 0.0f;
  __syncthreads();

  const int arow = mt * 16 + ln;  // row this lane supplies for A-operands

  for (int i = 0; i < D_; ++i) {
    // ---- fc1: h1 = relu(inp @ W1m[i]^T), K = 256 ------------------------
    const __bf16* w1i = w1m + (size_t)i * H_ * D_;
    v16bf a1[8];
    #pragma unroll
    for (int kt = 0; kt < 8; ++kt)
      a1[kt] = load_a_frag(s_inp + arow * D_ + kt * 32 + lh * 8);

    for (int j = 0; j < 16; ++j) {
      const int n = (ng * 16 + j) * 16 + ln;
      v8f acc = {};
      const __bf16* bp = w1i + (size_t)n * D_ + lh * 8;
      #pragma unroll
      for (int kt = 0; kt < 8; ++kt) {
        v16bf b = load_b_frag(bp + kt * 32);
        acc = __builtin_amdgcn_wmma_f32_16x16x32_bf16(
            false, a1[kt], false, b, (short)0, acc, false, false);
      }
      // C layout: lane half lh -> rows r+8*lh, N = ln
      #pragma unroll
      for (int r = 0; r < 8; ++r) {
        float h = acc[r];
        h = h > 0.0f ? h : 0.0f;
        s_h1[(mt * 16 + 8 * lh + r) * H_ + n] = (__bf16)h;
      }
    }
    __syncthreads();

    // ---- fc2 (+ fused fc3 partial): relu(h1 @ W2[i]^T + b2) . w3 --------
    const __bf16* w2i = w2 + (size_t)i * H_ * H_;
    const __bf16* ap  = s_h1 + arow * H_ + lh * 8;
    float p[8];
    #pragma unroll
    for (int r = 0; r < 8; ++r) p[r] = 0.0f;

    for (int j = 0; j < 16; ++j) {
      const int n = (ng * 16 + j) * 16 + ln;
      v8f acc = {};
      const __bf16* bp = w2i + (size_t)n * H_ + lh * 8;
      #pragma unroll
      for (int kt = 0; kt < 16; ++kt) {
        v16bf a = load_a_frag(ap + kt * 32);
        v16bf b = load_b_frag(bp + kt * 32);
        acc = __builtin_amdgcn_wmma_f32_16x16x32_bf16(
            false, a, false, b, (short)0, acc, false, false);
      }
      const float b2v = b2[(size_t)i * H_ + n];
      const float w3v = w3[(size_t)i * H_ + n];
      #pragma unroll
      for (int r = 0; r < 8; ++r) {
        float h = acc[r] + b2v;
        h = h > 0.0f ? h : 0.0f;
        p[r] = fmaf(h, w3v, p[r]);
      }
    }
    #pragma unroll
    for (int r = 0; r < 8; ++r)
      atomicAdd(&s_out[mt * 16 + 8 * lh + r], p[r]);
    __syncthreads();

    // ---- write column i: out, and into s_inp for later steps -----------
    if (tid < BT) {
      float v = s_out[tid] + b3[i];
      out[(size_t)(row0 + tid) * D_ + i] = v;
      s_inp[tid * D_ + i] = (__bf16)v;
      s_out[tid] = 0.0f;
    }
    __syncthreads();
  }
}

extern "C" void kernel_launch(void* const* d_in, const int* in_sizes, int n_in,
                              void* d_out, int out_size, void* d_ws, size_t ws_size,
                              hipStream_t stream) {
  const float* z    = (const float*)d_in[0];
  const float* W1   = (const float*)d_in[1];
  const float* W2   = (const float*)d_in[2];
  const float* b2   = (const float*)d_in[3];
  const float* W3   = (const float*)d_in[4];
  const float* b3   = (const float*)d_in[5];
  const int*   mask = (const int*)d_in[6];
  float* out = (float*)d_out;

  // Workspace layout (needs ~192 MiB): bf16 W1 (masked) then bf16 W2
  __bf16* w1m = (__bf16*)d_ws;
  __bf16* w2b = (__bf16*)((char*)d_ws + (size_t)D_ * H_ * D_ * sizeof(__bf16));

  const int n1 = D_ * H_ * D_;  // 33,554,432
  const int n2 = D_ * H_ * H_;  // 67,108,864
  cvt_w1_masked<<<2048, 256, 0, stream>>>(W1, mask, w1m, n1);
  cvt_plain    <<<4096, 256, 0, stream>>>(W2, w2b, n2);

  sparse_mlp_chain<<<B_ / BT, NTHR, 0, stream>>>(z, w1m, w2b, b2, W3, b3, out);
}
